// TreeModel_6725918785677
// MI455X (gfx1250) — matrix operand order repaired and checked by reading, not compiled, not run on hardware
//
#include <hip/hip_runtime.h>
#include <hip/hip_bf16.h>
#include <cstdint>
#include <cstddef>

typedef __attribute__((ext_vector_type(16))) _Float16 v16h;
typedef __attribute__((ext_vector_type(8)))  _Float16 v8h;
typedef __attribute__((ext_vector_type(8)))  float    v8f;

// ---- CDNA5 async global->LDS copy (ASYNCcnt-tracked, ISA §15.18.3) ----
// VDST = LDS byte offset (within wave's LDS alloc), VADDR = 64-bit global addr.
static __device__ __forceinline__ void async_copy_b128(const void* gsrc,
                                                       unsigned lds_off) {
  asm volatile("global_load_async_to_lds_b128 %0, %1, off"
               :
               : "v"(lds_off), "v"(gsrc)
               : "memory");
}
static __device__ __forceinline__ void wait_async0() {
  asm volatile("s_wait_asynccnt 0" ::: "memory");
}

// ---- WMMA fragment loaders (CDNA5 16x16x32 f16 layouts, ISA 7.12.2) ----
// A (16xK tile, row-major, leading dim ld):
//   lanes 0-15 hold M=lane, K = {0..7, 16..23}; lanes 16-31: K = {8..15, 24..31}
static __device__ __forceinline__ v16h load_fragA(const _Float16* base, int ld, int lane) {
  const int m = lane & 15, half = lane >> 4;
  const _Float16* p = base + (size_t)m * ld + half * 8;
  v8h lo = *reinterpret_cast<const v8h*>(p);
  v8h hi = *reinterpret_cast<const v8h*>(p + 16);
  v16h r;
#pragma unroll
  for (int i = 0; i < 8; ++i) { r[i] = lo[i]; r[i + 8] = hi[i]; }
  return r;
}
// B (Kx16 tile) loaded from pre-transposed W^T stored row-major [N][K]:
//   lanes 0-15 hold N=lane, K = {0..7, 16..23}; lanes 16-31: K = {8..15, 24..31}
static __device__ __forceinline__ v16h load_fragB(const _Float16* base, int ld, int lane) {
  const int n = lane & 15, half = lane >> 4;
  const _Float16* p = base + (size_t)n * ld + half * 8;
  v8h lo = *reinterpret_cast<const v8h*>(p);
  v8h hi = *reinterpret_cast<const v8h*>(p + 16);
  v16h r;
#pragma unroll
  for (int i = 0; i < 8; ++i) { r[i] = lo[i]; r[i + 8] = hi[i]; }
  return r;
}

static __device__ __forceinline__ v8f wmma16(v16h a, v16h b, v8f c) {
  return __builtin_amdgcn_wmma_f32_16x16x32_f16(false, a, false, b, (short)0, c,
                                                false, false);
}

static __device__ __forceinline__ float sigmoidf_(float x) {
  return 1.0f / (1.0f + __expf(-x));
}

// ============================================================================
// Fused GRU scan: one workgroup (8 wave32) owns a 16-row tile across all SEQ
// steps.  concat([x_t, h]) lives in LDS as f16; D = A*B with
// v_wmma_f32_16x16x32_f16;  gates applied in registers on the accumulator
// layout (VGPR v -> M = (lane>=16 ? 8 : 0)+v, N = lane&15).
//   WLDS=true : W^T matrices cached in LDS (sentence level, fits 320KB WGP LDS)
//   WLDS=false: W^T streamed from global/L2 (document level)
// All global->LDS staging uses the CDNA5 async path (no VGPR bounce).
// ============================================================================
template <int DIN, int H, int SEQ, bool WLDS, typename OutT>
__global__ __launch_bounds__(256) void gru_wmma(
    const _Float16* __restrict__ xseq,  // [Nrows, SEQ, DIN] f16
    const int* __restrict__ lens,       // [Nrows]
    const _Float16* __restrict__ WgT,   // [2H][DIN+H] f16 (transposed)
    const _Float16* __restrict__ WcT,   // [H][DIN+H]  f16 (transposed)
    const float* __restrict__ bg,       // [2H]
    const float* __restrict__ bc,       // [H]
    OutT* __restrict__ out,             // [Nrows, out_ld], writes H cols at base
    int out_ld, int reverse) {
  constexpr int CK = DIN + H;
  constexpr int NTG = (2 * H) / 16;   // gate GEMM n-tiles
  constexpr int NTC = H / 16;         // candidate GEMM n-tiles
  constexpr int TPW_G = NTG / 8;
  constexpr int TPW_C = NTC / 8;

  extern __shared__ __align__(16) char smem_raw[];
  const int tid = threadIdx.x;
  const int lane = tid & 31;
  const int wave = tid >> 5;
  const int rowbase = blockIdx.x * 16;

  const _Float16* wg_p;
  const _Float16* wc_p;
  _Float16* xa;  // [16][CK] concat(x_t, h) in f16
  if constexpr (WLDS) {
    _Float16* sWg = reinterpret_cast<_Float16*>(smem_raw);
    _Float16* sWc = sWg + 2 * H * CK;
    xa = sWc + H * CK;
    // async DMA weight fill: one b128 per element-group, no VGPR round-trip
    for (int i = tid * 8; i < 2 * H * CK; i += 256 * 8)
      async_copy_b128(WgT + i,
                      (unsigned)((const char*)(sWg + i) - smem_raw));
    for (int i = tid * 8; i < H * CK; i += 256 * 8)
      async_copy_b128(WcT + i,
                      (unsigned)((const char*)(sWc + i) - smem_raw));
    wg_p = sWg;
    wc_p = sWc;
  } else {
    wg_p = WgT;
    wc_p = WcT;
    xa = reinterpret_cast<_Float16*>(smem_raw);
  }
  float* ubuf = reinterpret_cast<float*>(xa + 16 * CK);  // [16][H] update gate
  float* h32  = ubuf + 16 * H;                           // [16][H] hidden (f32)
  int* slen   = reinterpret_cast<int*>(h32 + 16 * H);    // [16]

  if (tid < 16) slen[tid] = lens[rowbase + tid];
  for (int i = tid; i < 16 * H; i += 256) {
    h32[i] = 0.0f;
    xa[(i / H) * CK + DIN + (i % H)] = (_Float16)0.0f;
  }
  if constexpr (WLDS) wait_async0();
  __syncthreads();

  const v8f vz = {};
  for (int t = 0; t < SEQ; ++t) {
    // ---- stage x_t into xa[:, 0:DIN] via async copy (reverse folded in) ----
    {
      const int r = tid >> 4;
      const int len = slen[r];
      const int tt = reverse ? ((t < len) ? (len - 1 - t) : t) : t;
      const _Float16* src = xseq + ((size_t)(rowbase + r) * SEQ + tt) * DIN;
      const int c0 = (tid & 15) * (DIN / 16);
#pragma unroll
      for (int c = 0; c < DIN / 16; c += 8)
        async_copy_b128(src + c0 + c,
                        (unsigned)((const char*)(xa + r * CK + c0 + c) - smem_raw));
      wait_async0();
    }
    __syncthreads();

    // ---- gate GEMM: ru = [16,CK] @ Wg -> [16, 2H] ----
    v8f gacc[TPW_G];
#pragma unroll
    for (int i = 0; i < TPW_G; ++i) gacc[i] = vz;
#pragma unroll
    for (int k = 0; k < CK; k += 32) {
      v16h a = load_fragA(xa + k, CK, lane);
#pragma unroll
      for (int i = 0; i < TPW_G; ++i) {
        v16h b = load_fragB(wg_p + (size_t)(wave * TPW_G + i) * 16 * CK + k, CK, lane);
        gacc[i] = wmma16(a, b, gacc[i]);
      }
    }
    __syncthreads();
    // sigmoid gates in-register; r-gate waves write r*h into xa, u -> ubuf
    {
      const int ncol0 = lane & 15;
      const int mbase = (lane >> 4) * 8;
#pragma unroll
      for (int i = 0; i < TPW_G; ++i) {
        const int ncol = (wave * TPW_G + i) * 16 + ncol0;
        const float bias = bg[ncol];
#pragma unroll
        for (int v = 0; v < 8; ++v) {
          const int m = mbase + v;
          const float g = sigmoidf_(gacc[i][v] + bias);
          if (ncol < H) {
            xa[m * CK + DIN + ncol] = (_Float16)(g * h32[m * H + ncol]);
          } else {
            ubuf[m * H + (ncol - H)] = g;
          }
        }
      }
    }
    __syncthreads();

    // ---- candidate GEMM: c = [16,CK(x, r*h)] @ Wc -> [16, H] ----
    v8f cacc[TPW_C];
#pragma unroll
    for (int i = 0; i < TPW_C; ++i) cacc[i] = vz;
#pragma unroll
    for (int k = 0; k < CK; k += 32) {
      v16h a = load_fragA(xa + k, CK, lane);
#pragma unroll
      for (int i = 0; i < TPW_C; ++i) {
        v16h b = load_fragB(wc_p + (size_t)(wave * TPW_C + i) * 16 * CK + k, CK, lane);
        cacc[i] = wmma16(a, b, cacc[i]);
      }
    }
    __syncthreads();
    // ---- h' = u*h + (1-u)*tanh(c), masked by t < len ----
    {
      const int ncol0 = lane & 15;
      const int mbase = (lane >> 4) * 8;
#pragma unroll
      for (int i = 0; i < TPW_C; ++i) {
        const int ncol = (wave * TPW_C + i) * 16 + ncol0;
        const float bias = bc[ncol];
#pragma unroll
        for (int v = 0; v < 8; ++v) {
          const int m = mbase + v;
          const float c = tanhf(cacc[i][v] + bias);
          const float u = ubuf[m * H + ncol];
          const float hold = h32[m * H + ncol];
          float hn = u * hold + (1.0f - u) * c;
          if (t >= slen[m]) hn = hold;
          h32[m * H + ncol] = hn;
          xa[m * CK + DIN + ncol] = (_Float16)hn;
        }
      }
    }
    __syncthreads();
  }

  for (int i = tid; i < 16 * H; i += 256) {
    const int m = i / H, n = i % H;
    out[(size_t)(rowbase + m) * out_ld + n] = (OutT)h32[i];
  }
}

// ---- embedding gather + f32->f16 convert: words[row][w][d] = emb[X[row][w]][d]
__global__ __launch_bounds__(256) void gather_embed(const int* __restrict__ X,
                                                    const float* __restrict__ emb,
                                                    _Float16* __restrict__ words,
                                                    int total, int W, int D) {
  int i = blockIdx.x * blockDim.x + threadIdx.x;
  if (i < total) {
    int d = i % D;
    int w = (i / D) % W;
    int row = i / (D * W);
    int id = X[row * W + w];
    words[i] = (_Float16)emb[(size_t)id * D + d];
  }
}

// ---- one-time weight prep: f32 [K][N] row-major -> f16 [N][K] (transposed)
__global__ __launch_bounds__(256) void convert_transpose(const float* __restrict__ src,
                                                         _Float16* __restrict__ dst,
                                                         int K, int N) {
  int i = blockIdx.x * blockDim.x + threadIdx.x;
  if (i < K * N) {
    int k = i / N, n = i % N;
    dst[(size_t)n * K + k] = (_Float16)src[i];
  }
}

// ---- MLP head: logits = relu(doc @ W1 + b1) @ W2 + b2 (tiny, fp32 VALU) ----
__global__ __launch_bounds__(256) void mlp_head(const float* __restrict__ doc,
                                                const float* __restrict__ W1,
                                                const float* __restrict__ b1,
                                                const float* __restrict__ W2,
                                                const float* __restrict__ b2,
                                                float* __restrict__ out) {
  __shared__ float hid[256];
  const int row = blockIdx.x;
  const int j = threadIdx.x;
  const float* dr = doc + (size_t)row * 512;
  float acc = b1[j];
  for (int k = 0; k < 512; ++k) acc += dr[k] * W1[k * 256 + j];
  hid[j] = fmaxf(acc, 0.0f);
  __syncthreads();
  if (j < 5) {
    float a = b2[j];
    for (int k = 0; k < 256; ++k) a += hid[k] * W2[k * 5 + j];
    out[row * 5 + j] = a;
  }
}

extern "C" void kernel_launch(void* const* d_in, const int* in_sizes, int n_in,
                              void* d_out, int out_size, void* d_ws, size_t ws_size,
                              hipStream_t stream) {
  (void)in_sizes; (void)n_in; (void)out_size; (void)ws_size;
  const int* X    = (const int*)d_in[0];
  const int* L    = (const int*)d_in[1];
  const int* L2   = (const int*)d_in[2];
  const float* emb   = (const float*)d_in[3];
  const float* sWg_f = (const float*)d_in[4];
  const float* sbg_f = (const float*)d_in[5];
  const float* sWc_f = (const float*)d_in[6];
  const float* sbc_f = (const float*)d_in[7];
  const float* sWg_b = (const float*)d_in[8];
  const float* sbg_b = (const float*)d_in[9];
  const float* sWc_b = (const float*)d_in[10];
  const float* sbc_b = (const float*)d_in[11];
  const float* dWg_f = (const float*)d_in[12];
  const float* dbg_f = (const float*)d_in[13];
  const float* dWc_f = (const float*)d_in[14];
  const float* dbc_f = (const float*)d_in[15];
  const float* dWg_b = (const float*)d_in[16];
  const float* dbg_b = (const float*)d_in[17];
  const float* dWc_b = (const float*)d_in[18];
  const float* dbc_b = (const float*)d_in[19];
  const float* W1 = (const float*)d_in[20];
  const float* b1 = (const float*)d_in[21];
  const float* W2 = (const float*)d_in[22];
  const float* b2 = (const float*)d_in[23];
  float* out = (float*)d_out;

  constexpr int B = 64, S = 32, W = 64, D = 128;

  size_t off = 0;
  auto take = [&](size_t bytes) {
    void* p = (char*)d_ws + off;
    off += (bytes + 255) & ~(size_t)255;
    return p;
  };
  _Float16* words  = (_Float16*)take((size_t)B * S * W * D * 2);  // 33.5 MB
  _Float16* sWgT_f = (_Float16*)take(256 * 256 * 2);
  _Float16* sWcT_f = (_Float16*)take(128 * 256 * 2);
  _Float16* sWgT_b = (_Float16*)take(256 * 256 * 2);
  _Float16* sWcT_b = (_Float16*)take(128 * 256 * 2);
  _Float16* dWgT_f = (_Float16*)take(512 * 512 * 2);
  _Float16* dWcT_f = (_Float16*)take(256 * 512 * 2);
  _Float16* dWgT_b = (_Float16*)take(512 * 512 * 2);
  _Float16* dWcT_b = (_Float16*)take(256 * 512 * 2);
  _Float16* sent   = (_Float16*)take((size_t)B * S * 256 * 2);
  float* docv      = (float*)take((size_t)B * 512 * 4);

  const int total_words = B * S * W * D;  // 16,777,216
  gather_embed<<<(total_words + 255) / 256, 256, 0, stream>>>(X, emb, words,
                                                              total_words, W, D);

  auto convT = [&](const float* s, _Float16* dT, int K, int N) {
    convert_transpose<<<((K * N) + 255) / 256, 256, 0, stream>>>(s, dT, K, N);
  };
  convT(sWg_f, sWgT_f, 256, 256);
  convT(sWc_f, sWcT_f, 256, 128);
  convT(sWg_b, sWgT_b, 256, 256);
  convT(sWc_b, sWcT_b, 256, 128);
  convT(dWg_f, dWgT_f, 512, 512);
  convT(dWc_f, dWcT_f, 512, 256);
  convT(dWg_b, dWgT_b, 512, 512);
  convT(dWc_b, dWcT_b, 512, 256);

  // Sentence bi-GRU: 2048 rows, DIN=128, H=128, T=64; weights LDS-resident.
  constexpr size_t lds_sent =
      (size_t)(2 * 128 * 256 + 128 * 256 + 16 * 256) * 2 +  // Wg^T, Wc^T, xa
      (size_t)(16 * 128 * 2) * 4 + 16 * 4;                   // ubuf, h32, slen
  gru_wmma<128, 128, 64, true, _Float16>
      <<<dim3((B * S) / 16), 256, lds_sent, stream>>>(
          words, L, sWgT_f, sWcT_f, sbg_f, sbc_f, sent, 256, 0);
  gru_wmma<128, 128, 64, true, _Float16>
      <<<dim3((B * S) / 16), 256, lds_sent, stream>>>(
          words, L, sWgT_b, sWcT_b, sbg_b, sbc_b, sent + 128, 256, 1);

  // Document bi-GRU: 64 rows, DIN=256, H=256, T=32; weights stream from L2.
  constexpr size_t lds_doc =
      (size_t)(16 * 512) * 2 + (size_t)(16 * 256 * 2) * 4 + 16 * 4;
  gru_wmma<256, 256, 32, false, float>
      <<<dim3(B / 16), 256, lds_doc, stream>>>(
          sent, L2, dWgT_f, dWcT_f, dbg_f, dbc_f, docv, 512, 0);
  gru_wmma<256, 256, 32, false, float>
      <<<dim3(B / 16), 256, lds_doc, stream>>>(
          sent, L2, dWgT_b, dWcT_b, dbg_b, dbc_b, docv + 256, 512, 1);

  mlp_head<<<B, 256, 0, stream>>>(docv, W1, b1, W2, b2, out);
}
